// sLSTMLayer_37048387895686
// MI455X (gfx1250) — matrix operand order repaired and checked by reading, not compiled, not run on hardware
//
#include <hip/hip_runtime.h>
#include <hip/hip_bf16.h>

#define BD  8
#define SD  2048
#define DD  1024
#define NHD 4
#define DHD 256
#define KW  4

typedef __attribute__((ext_vector_type(16))) __bf16 v16bf;
typedef __attribute__((ext_vector_type(8)))  float  v8f;
typedef __attribute__((ext_vector_type(4)))  unsigned int v4u;
typedef __attribute__((ext_vector_type(4)))  unsigned int u32x4;
typedef __attribute__((ext_vector_type(8)))  int i32x8;
typedef __attribute__((ext_vector_type(4)))  int i32x4;

union FragU { v16bf v; v4u q[2]; };

__device__ __forceinline__ unsigned short f2bf(float f) {
  __hip_bfloat16 h = __float2bfloat16(f);
  return *reinterpret_cast<unsigned short*>(&h);
}
__device__ __forceinline__ float bf2f(unsigned short u) {
  unsigned v = ((unsigned)u) << 16;
  return __builtin_bit_cast(float, v);
}

// ---------------- Kernel 1: depthwise causal conv + SiLU, emit bf16 x and xc --
__global__ void __launch_bounds__(256) k_conv_silu(
    const float* __restrict__ x, const float* __restrict__ cw,
    const float* __restrict__ cb, unsigned short* __restrict__ xcb,
    unsigned short* __restrict__ xb) {
  int idx = blockIdx.x * 256 + threadIdx.x;          // over B*S*D = 2^24
  int d  = idx & (DD - 1);
  int s  = (idx >> 10) & (SD - 1);
  int b  = idx >> 21;
  float xv  = x[idx];
  float acc = cb[d];
#pragma unroll
  for (int j = 0; j < KW; ++j) {
    int ss = s + j - (KW - 1);
    float xj = (ss >= 0) ? x[((size_t)b * SD + ss) * DD + d] : 0.f;
    acc = fmaf(xj, cw[j * DD + d], acc);
  }
  float sl = acc / (1.f + __expf(-acc));             // SiLU
  xcb[idx] = f2bf(sl);
  xb[idx]  = f2bf(xv);
}

// ---------------- Kernel 2: f32 -> bf16 weight conversion -------------------
__global__ void k_cvt(const float* __restrict__ src, unsigned short* __restrict__ dst, int n) {
  int i = blockIdx.x * 256 + threadIdx.x;
  if (i < n) dst[i] = f2bf(src[i]);
}

// ---------------- Kernel 3: Wx = gate pre-activations via bf16 WMMA ----------
// C[token, (g,h,kout)] = sum_d A[token,d] * w[g,h,kout,d]
// wx layout (bf16): [h][s][g][b][k]  (scan streams contiguous 16KB per step)
__global__ void __launch_bounds__(256) k_gemm_wx(
    const unsigned short* __restrict__ xcb, const unsigned short* __restrict__ xb,
    const unsigned short* __restrict__ wbuf, unsigned short* __restrict__ wx) {
  const int lane   = threadIdx.x & 31;
  const int wave   = threadIdx.x >> 5;
  const int tileId = blockIdx.x * 8 + wave;          // 262144 tiles total
  const int mt   = tileId >> 8;                      // 1024 M-tiles (16 tokens)
  const int rest = tileId & 255;
  const int g  = rest >> 6;
  const int h  = (rest >> 4) & 3;
  const int nt = rest & 15;
  const int mn = lane & 15;                          // M (A) / N (B,C) sub-index
  const int kh = lane >> 4;                          // K-half selector

  const unsigned short* A    = (g < 2) ? xcb : xb;   // i,f use conv path; z,o raw x
  const unsigned short* arow = A + ((size_t)(mt * 16 + mn)) * DD + h * DHD;
  const unsigned short* brow = wbuf +
      (((size_t)g * NHD + h) * DHD + nt * 16 + mn) * DHD + kh * 8;

  v8f acc = {0.f,0.f,0.f,0.f,0.f,0.f,0.f,0.f};
#pragma unroll
  for (int kb = 0; kb < 8; ++kb) {
    FragU a, b;
    const int ks = kb * 32;
    a.q[0] = *(const v4u*)(arow + ks + kh * 8);      // A: K = kh*8 + e  (e<8)
    a.q[1] = *(const v4u*)(arow + ks + kh * 8 + 16); //    K = 16 + kh*8 + (e-8)
    b.q[0] = *(const v4u*)(brow + ks);               // B: K = kh*8 + e (e<8), 16 + kh*8 + (e-8)
    b.q[1] = *(const v4u*)(brow + ks + 16);
    acc = __builtin_amdgcn_wmma_f32_16x16x32_bf16(false, a.v, false, b.v,
                                                  (short)0, acc, false, false);
  }
  const int col = nt * 16 + mn;
#pragma unroll
  for (int r = 0; r < 8; ++r) {
    int m     = r + 8 * kh;                          // C/D row layout
    int token = mt * 16 + m;
    int bb = token >> 11;
    int ss = token & (SD - 1);
    wx[(((size_t)h * SD + ss) * 4 + g) * (BD * DHD) + bb * DHD + col] = f2bf(acc[r]);
  }
}

// ---------------- Kernel 4: sequential sLSTM scan + fused GroupNorm ----------
// One block per head. 16 waves; wave w owns output columns [16w,16w+16) for all
// 4 gates. Per step: 32 WMMAs (M=16 pad of B=8, K=256) with h_{t-1} from LDS.
// wx_t (16KB bf16) is double-buffered in LDS via TDM (TENSORcnt) or per-lane
// async-to-LDS loads (ASYNCcnt), issued one step ahead.

#define WXN (4 * BD * DHD)     /* 8192 bf16 elements = 16KB per step */

#if __has_builtin(__builtin_amdgcn_tensor_load_to_lds)
#define USE_TDM 1
#else
#define USE_TDM 0
#endif

__device__ __forceinline__ void stage_wx(const unsigned short* gsrc,
                                         unsigned short* lds_dst, int tid) {
#if USE_TDM
  // one TDM descriptor: 1-D tile, data_size=2B, 8192 elements
  unsigned long long ga = (unsigned long long)(size_t)gsrc;
  unsigned lds_addr = (unsigned)(size_t)lds_dst;     // low 32 bits = LDS byte addr
  u32x4 g0 = { 1u,                                   // count=1, user-mode
               lds_addr,
               (unsigned)(ga & 0xffffffffu),
               (unsigned)((ga >> 32) & 0x01ffffffu) | (2u << 30) };  // type=2
  i32x8 g1 = { (int)(1u << 16),                      // data_size=1 (2 bytes)
               (int)(((unsigned)WXN & 0xffffu) << 16),  // tensor_dim0[15:0]
               (int)(((unsigned)WXN >> 16) | (1u << 16)), // tensor_dim0 hi | tensor_dim1=1
               (int)(((unsigned)WXN & 0xffffu) << 16),  // tile_dim0 = 8192
               1,                                    // tile_dim1 = 1
               WXN,                                  // tensor_dim0_stride
               0, 0 };
  i32x4 gz = { 0, 0, 0, 0 };
#if defined(__clang_major__) && (__clang_major__ >= 23)
  i32x8 gz8 = { 0, 0, 0, 0, 0, 0, 0, 0 };
  __builtin_amdgcn_tensor_load_to_lds(g0, g1, gz, gz, gz8, 0);
#else
  __builtin_amdgcn_tensor_load_to_lds(g0, g1, gz, gz, 0);
#endif
  (void)tid;
#else
  // per-lane async copy: 1024 chunks of 16B, 2 per thread
  unsigned ldsbase = (unsigned)(size_t)lds_dst;
  for (int i = tid; i < WXN / 8; i += 512) {
    asm volatile("global_load_async_to_lds_b128 %0, %1, off"
                 :: "v"(ldsbase + (unsigned)i * 16u), "v"(gsrc + i * 8)
                 : "memory");
  }
#endif
}

__device__ __forceinline__ void wait_stage() {
#if USE_TDM
  __builtin_amdgcn_s_wait_tensorcnt(0);
#else
#if __has_builtin(__builtin_amdgcn_s_wait_asynccnt)
  __builtin_amdgcn_s_wait_asynccnt(0);
#else
  asm volatile("s_wait_asynccnt 0x0" ::: "memory");
#endif
#endif
}

__global__ void __launch_bounds__(512) k_scan(
    const unsigned short* __restrict__ wx, const unsigned short* __restrict__ rbuf,
    const float* __restrict__ rb, const float* __restrict__ gn_w,
    float* __restrict__ out) {
  const int h    = blockIdx.x;
  const int tid  = threadIdx.x;
  const int lane = tid & 31;
  const int wave = tid >> 5;                         // = n-tile
  const int n    = lane & 15;
  const int kh   = lane >> 4;
  const int k    = wave * 16 + n;                    // output column in [0,256)

  __shared__ __align__(16) unsigned short wxs2[2][WXN];        // 2 x 16KB staging
  __shared__ __align__(16) unsigned short hbuf[16 * DHD];      // bf16 h (padded M=16)
  __shared__ float red[16];                                    // sum / sumsq per b

  for (int i = tid; i < 16 * DHD; i += 512) hbuf[i] = 0;       // h_0 = 0 (+pad rows)

  float c_s[8], n_s[8], m_s[8], hn[8];
#pragma unroll
  for (int r = 0; r < 8; ++r) { c_s[r] = 0.f; n_s[r] = 0.f; m_s[r] = 0.f; }

  float rbv[4];
#pragma unroll
  for (int g = 0; g < 4; ++g) rbv[g] = rb[(g * NHD + h) * DHD + k];
  const float gnv = gn_w[h * DHD + k];

  const size_t gstride = (size_t)NHD * DHD * DHD;
  const unsigned short* brow0 = rbuf + ((size_t)h * DHD + k) * DHD + kh * 8;
  const unsigned short* wx_h  = wx + (size_t)h * SD * WXN;

  // prologue: stage step 0 into buffer 0
#if USE_TDM
  if (wave == 0) stage_wx(wx_h, &wxs2[0][0], tid);
#else
  stage_wx(wx_h, &wxs2[0][0], tid);
#endif
  __syncthreads();

  int cur = 0;
  for (int t = 0; t < SD; ++t) {
    // ---- wait for this step's staged wx, then kick off next step's DMA ----
#if USE_TDM
    if (wave == 0) wait_stage();
#else
    wait_stage();
#endif
    __syncthreads();                                  // (A) wxs2[cur] visible to all

    if (t + 1 < SD) {
      const unsigned short* nsrc = wx_h + (size_t)(t + 1) * WXN;
#if USE_TDM
      if (wave == 0) stage_wx(nsrc, &wxs2[cur ^ 1][0], tid);
#else
      stage_wx(nsrc, &wxs2[cur ^ 1][0], tid);
#endif
    }
    if (tid == 0 && t + 8 < SD)                       // warm GL2 ahead of the DMA
      __builtin_prefetch((const void*)(wx_h + (size_t)(t + 8) * WXN), 0, 1);

    if (tid < 16) red[tid] = 0.f;

    // ---- rec = h_{t-1} @ R_g^T via WMMA ----
    v8f acc[4];
#pragma unroll
    for (int g = 0; g < 4; ++g) {
      v8f a0 = {0.f,0.f,0.f,0.f,0.f,0.f,0.f,0.f};
      const unsigned short* brow = brow0 + (size_t)g * gstride;
#pragma unroll
      for (int kb = 0; kb < 8; ++kb) {
        FragU af, bf_;
        const int ks = kb * 32;
        af.q[0] = *(const v4u*)(&hbuf[(lane & 15) * DHD + ks + kh * 8]);
        af.q[1] = *(const v4u*)(&hbuf[(lane & 15) * DHD + ks + kh * 8 + 16]);
        bf_.q[0] = *(const v4u*)(brow + ks);
        bf_.q[1] = *(const v4u*)(brow + ks + 16);
        a0 = __builtin_amdgcn_wmma_f32_16x16x32_bf16(false, af.v, false, bf_.v,
                                                     (short)0, a0, false, false);
      }
      acc[g] = a0;
    }

    // ---- pointwise sLSTM gating (lanes 0..15 hold valid b=0..7 rows) ----
#pragma unroll
    for (int r = 0; r < 8; ++r) hn[r] = 0.f;
    if (lane < 16) {
#pragma unroll
      for (int r = 0; r < 8; ++r) {
        float ir  = acc[0][r] + bf2f(wxs2[cur][(0 * BD + r) * DHD + k]) + rbv[0];
        float fr  = acc[1][r] + bf2f(wxs2[cur][(1 * BD + r) * DHD + k]) + rbv[1];
        float zr  = acc[2][r] + bf2f(wxs2[cur][(2 * BD + r) * DHD + k]) + rbv[2];
        float orr = acc[3][r] + bf2f(wxs2[cur][(3 * BD + r) * DHD + k]) + rbv[3];
        float ls = (fr >= 0.f) ? -log1pf(__expf(-fr)) : (fr - log1pf(__expf(fr)));
        float lm = m_s[r] + ls;
        float mn = fmaxf(ir, lm);
        float ig = __expf(ir - mn);
        float fg = __expf(lm - mn);
        float cn = fg * c_s[r] + ig * tanhf(zr);
        float nn = fg * n_s[r] + ig;
        float hv = (1.f / (1.f + __expf(-orr))) * cn / nn;
        c_s[r] = cn; n_s[r] = nn; m_s[r] = mn; hn[r] = hv;
      }
    }
    __syncthreads();                                  // (B) all hbuf reads done

    if (lane < 16) {
#pragma unroll
      for (int r = 0; r < 8; ++r) hbuf[r * DHD + k] = f2bf(hn[r]);
    }
    // ---- GroupNorm partial reduce: over 16 n-lanes in wave, then LDS atomics
#pragma unroll
    for (int r = 0; r < 8; ++r) {
      float v = hn[r], v2 = hn[r] * hn[r];
#pragma unroll
      for (int off = 8; off >= 1; off >>= 1) {
        v  += __shfl_xor(v,  off, 16);
        v2 += __shfl_xor(v2, off, 16);
      }
      if (lane == 0) { atomicAdd(&red[r], v); atomicAdd(&red[8 + r], v2); }
    }
    __syncthreads();                                  // (C) red complete

    if (lane < 16) {
#pragma unroll
      for (int r = 0; r < 8; ++r) {
        float mu   = red[r] * (1.f / DHD);
        float var  = red[8 + r] * (1.f / DHD) - mu * mu;
        float rstd = rsqrtf(var + 1e-5f);
        out[((size_t)r * SD + t) * DD + h * DHD + k] = (hn[r] - mu) * rstd * gnv;
      }
    }
    cur ^= 1;
    // next iteration's barrier (A) protects staging overwrite and red re-zero
  }
}

extern "C" void kernel_launch(void* const* d_in, const int* in_sizes, int n_in,
                              void* d_out, int out_size, void* d_ws, size_t ws_size,
                              hipStream_t stream) {
  const float* x   = (const float*)d_in[0];
  const float* cw  = (const float*)d_in[1];
  const float* cb  = (const float*)d_in[2];
  const float* w_i = (const float*)d_in[3];
  const float* w_f = (const float*)d_in[4];
  const float* w_z = (const float*)d_in[5];
  const float* w_o = (const float*)d_in[6];
  const float* R   = (const float*)d_in[7];
  const float* rb  = (const float*)d_in[8];
  const float* gnw = (const float*)d_in[9];
  float* out = (float*)d_out;

  char* ws = (char*)d_ws;
  unsigned short* xb   = (unsigned short*)(ws);                 // 32 MB
  unsigned short* xcb  = (unsigned short*)(ws + (33u << 20));   // 32 MB
  unsigned short* wbuf = (unsigned short*)(ws + (66u << 20));   // 2 MB
  unsigned short* rbuf = (unsigned short*)(ws + (69u << 20));   // 2 MB
  unsigned short* wx   = (unsigned short*)(ws + (72u << 20));   // 128 MB (bf16)

  k_conv_silu<<<(BD * SD * DD) / 256, 256, 0, stream>>>(x, cw, cb, xcb, xb);

  const int wn = NHD * DHD * DHD;                               // 262144
  k_cvt<<<wn / 256, 256, 0, stream>>>(w_i, wbuf + 0 * (size_t)wn, wn);
  k_cvt<<<wn / 256, 256, 0, stream>>>(w_f, wbuf + 1 * (size_t)wn, wn);
  k_cvt<<<wn / 256, 256, 0, stream>>>(w_z, wbuf + 2 * (size_t)wn, wn);
  k_cvt<<<wn / 256, 256, 0, stream>>>(w_o, wbuf + 3 * (size_t)wn, wn);
  k_cvt<<<4 * wn / 256, 256, 0, stream>>>(R, rbuf, 4 * wn);

  k_gemm_wx<<<32768, 256, 0, stream>>>(xcb, xb, wbuf, wx);      // 262144 tiles / 8 waves
  k_scan<<<NHD, 512, 0, stream>>>(wx, rbuf, rb, gnw, out);
}
